// SVD_5257039970656
// MI455X (gfx1250) — compile-verified, matrix operand order checked by reading, not run on hardware
//
#include <hip/hip_runtime.h>
#include <math.h>

#define Bn 32
#define Nn 2048
#define Dn 512
#define BLOCK_M 64          // src rows per workgroup (4 waves x 16 rows)
#define TILE_N 32           // tgt rows per iteration (2 WMMA column tiles)
#define NTILES (Nn / TILE_N)
#define THREADS 128
#define QSTR (Dn + 8)       // LDS row stride in bf16 elems (1040 B; 1040/4 dwords == 4 mod 64 -> conflict-free)
#define SCALE_F 0.044194173824159216f  // 1/sqrt(512)

typedef __attribute__((ext_vector_type(16))) __bf16 v16bf;
typedef __attribute__((ext_vector_type(8)))  __bf16 v8bf;
typedef __attribute__((ext_vector_type(8)))  float  v8f;

union bffrag { v16bf v; v8bf h[2]; };

static __device__ __forceinline__ unsigned short f2bfraw(float f) {
  unsigned u = __builtin_bit_cast(unsigned, f);
  unsigned r = u + 0x7FFFu + ((u >> 16) & 1u);   // round-to-nearest-even
  return (unsigned short)(r >> 16);
}
static __device__ __forceinline__ unsigned pack2(float a, float b) {
  return (unsigned)f2bfraw(a) | ((unsigned)f2bfraw(b) << 16);
}
static __device__ __forceinline__ float redmax16(float v) {
  v = fmaxf(v, __shfl_xor(v, 1));
  v = fmaxf(v, __shfl_xor(v, 2));
  v = fmaxf(v, __shfl_xor(v, 4));
  v = fmaxf(v, __shfl_xor(v, 8));
  return v;
}
static __device__ __forceinline__ float redsum16(float v) {
  v += __shfl_xor(v, 1);
  v += __shfl_xor(v, 2);
  v += __shfl_xor(v, 4);
  v += __shfl_xor(v, 8);
  return v;
}
// LDS byte offset of a __shared__ object (generic ptr low 32 bits = LDS offset)
static __device__ __forceinline__ unsigned lds_off(const void* p) {
  return (unsigned)(unsigned long long)p;
}

// ---------------------------------------------------------------------------
// Pass 0 (fast path): elementwise f32 -> bf16 for the embeddings.
// ---------------------------------------------------------------------------
__global__ __launch_bounds__(256)
void convert_f32_bf16(const float* __restrict__ in, uint2* __restrict__ out, int nquads) {
  int i = blockIdx.x * 256 + threadIdx.x;
  if (i < nquads) {
    float4 x = ((const float4*)in)[i];
    uint2 u;
    u.x = pack2(x.x, x.y);
    u.y = pack2(x.z, x.w);
    out[i] = u;
  }
}

// ---------------------------------------------------------------------------
// Fast path: bf16 operands in memory; stage via global_load_async_to_lds_b128
// (ASYNCcnt DMA, double-buffered K tiles), WMMA scores + online softmax with
// V = tgt positions (3 cols). Writes matched (B, N, 3).
// ---------------------------------------------------------------------------
__global__ __launch_bounds__(THREADS)
void pointer_attention_async(const __bf16* __restrict__ qbf,
                             const __bf16* __restrict__ kbf,
                             const float* __restrict__ tgt,
                             float* __restrict__ matched) {
  __shared__ __bf16 sQ[BLOCK_M][QSTR];          // 66560 B
  __shared__ __bf16 sKd[2][TILE_N][QSTR];       // 66560 B (double buffer)

  const int b    = blockIdx.y;
  const int m0   = blockIdx.x * BLOCK_M;
  const int tid  = threadIdx.x;
  const int lane = tid & 31;
  const int wave = tid >> 5;
  const int half = lane >> 4;   // K-split group per ISA 16-bit A/B layout
  const int idx  = lane & 15;   // row (A) / column (B,C) within tile

  const unsigned sQ_base = lds_off(&sQ[0][0]);
  const unsigned sK_base0 = lds_off(&sKd[0][0][0]);
  const unsigned sK_base1 = lds_off(&sKd[1][0][0]);

  // ---- async stage Q block: 64 rows x 1024 B = 4096 16B chunks ----
  {
    const unsigned long long qg =
        (unsigned long long)(qbf + ((size_t)b * Nn + m0) * Dn);
    #pragma unroll 4
    for (int i = 0; i < 32; ++i) {
      int c = i * THREADS + tid;
      int row = c >> 6, col = c & 63;          // 64 chunks per row
      unsigned lds = sQ_base + (unsigned)(row * (QSTR * 2) + col * 16);
      unsigned long long g = qg + ((unsigned long long)row * (Dn * 2) + col * 16);
      asm volatile("global_load_async_to_lds_b128 %0, %1, off"
                   :: "v"(lds), "v"(g) : "memory");
    }
  }

  auto issue_k_tile = [&](int n0, unsigned kbase) {
    const unsigned long long kg =
        (unsigned long long)(kbf + ((size_t)b * Nn + n0) * Dn);
    #pragma unroll
    for (int i = 0; i < 16; ++i) {
      int c = i * THREADS + tid;               // 2048 chunks: 32 rows x 64
      int row = c >> 6, col = c & 63;
      unsigned lds = kbase + (unsigned)(row * (QSTR * 2) + col * 16);
      unsigned long long g = kg + ((unsigned long long)row * (Dn * 2) + col * 16);
      asm volatile("global_load_async_to_lds_b128 %0, %1, off"
                   :: "v"(lds), "v"(g) : "memory");
    }
  };

  issue_k_tile(0, sK_base0);   // prologue: tile 0 -> buffer 0

  float Mx[8], Ls[8], Ax[8], Ay[8], Az[8];
  #pragma unroll
  for (int j = 0; j < 8; ++j) { Mx[j] = -3.0e38f; Ls[j] = 0.f; Ax[j] = Ay[j] = Az[j] = 0.f; }

  const __bf16* qrow = &sQ[wave * 16 + idx][0];

  for (int kt = 0; kt < NTILES; ++kt) {
    // issue next tile into the other buffer, then retire current tile's DMA
    if (kt + 1 < NTILES) {
      issue_k_tile((kt + 1) * TILE_N, ((kt + 1) & 1) ? sK_base1 : sK_base0);
      asm volatile("s_wait_asynccnt 0x10" ::: "memory");   // keep 16 (next tile) in flight
    } else {
      asm volatile("s_wait_asynccnt 0x0" ::: "memory");
    }
    __syncthreads();   // all waves' DMA for this tile has landed in LDS

    const __bf16* krow0 = &sKd[kt & 1][idx][0];
    const __bf16* krow1 = &sKd[kt & 1][16 + idx][0];

    // ---- scores: two 16x16 tiles via v_wmma_f32_16x16x32_bf16 ----
    v8f c0 = {};
    v8f c1 = {};
    #pragma unroll
    for (int kc = 0; kc < Dn / 32; ++kc) {
      const int k0 = kc * 32 + half * 8;  // lanes<16: K{0..7,16..23}; lanes>=16: K{8..15,24..31}
      bffrag a, b0, b1;
      a.h[0]  = *(const v8bf*)(qrow  + k0);
      a.h[1]  = *(const v8bf*)(qrow  + k0 + 16);
      b0.h[0] = *(const v8bf*)(krow0 + k0);
      b0.h[1] = *(const v8bf*)(krow0 + k0 + 16);
      b1.h[0] = *(const v8bf*)(krow1 + k0);
      b1.h[1] = *(const v8bf*)(krow1 + k0 + 16);
      c0 = __builtin_amdgcn_wmma_f32_16x16x32_bf16(false, a.v, false, b0.v, (short)0, c0, false, false);
      c1 = __builtin_amdgcn_wmma_f32_16x16x32_bf16(false, a.v, false, b1.v, (short)0, c1, false, false);
    }

    // ---- online softmax + P@V (V read straight from global; 384 B/tile, L2-hot)
    const float* vp0 = tgt + ((size_t)b * Nn + kt * TILE_N + idx) * 3;
    const float v00 = vp0[0],  v01 = vp0[1],  v02 = vp0[2];
    const float v10 = vp0[48], v11 = vp0[49], v12 = vp0[50];   // +16 rows * 3
    #pragma unroll
    for (int j = 0; j < 8; ++j) {
      float s0 = c0[j] * SCALE_F;
      float s1 = c1[j] * SCALE_F;
      float mx   = redmax16(fmaxf(s0, s1));
      float mnew = fmaxf(Mx[j], mx);
      float fscl = __expf(Mx[j] - mnew);
      float p0 = __expf(s0 - mnew);
      float p1 = __expf(s1 - mnew);
      Ls[j] = Ls[j] * fscl + p0 + p1;
      Ax[j] = Ax[j] * fscl + p0 * v00 + p1 * v10;
      Ay[j] = Ay[j] * fscl + p0 * v01 + p1 * v11;
      Az[j] = Az[j] * fscl + p0 * v02 + p1 * v12;
      Mx[j] = mnew;
    }
    __syncthreads();   // everyone done reading sKd[kt&1] before it is DMA'd again
  }

  #pragma unroll
  for (int j = 0; j < 8; ++j) {
    Ls[j] = redsum16(Ls[j]);
    Ax[j] = redsum16(Ax[j]);
    Ay[j] = redsum16(Ay[j]);
    Az[j] = redsum16(Az[j]);
  }
  if (idx == 0) {
    #pragma unroll
    for (int j = 0; j < 8; ++j) {
      int row = m0 + wave * 16 + j + 8 * half;
      float inv = 1.0f / Ls[j];
      float* mp = matched + ((size_t)b * Nn + row) * 3;
      mp[0] = Ax[j] * inv;
      mp[1] = Ay[j] * inv;
      mp[2] = Az[j] * inv;
    }
  }
}

// ---------------------------------------------------------------------------
// Fallback (small workspace): round-1 fused kernel with on-the-fly conversion.
// ---------------------------------------------------------------------------
__global__ __launch_bounds__(THREADS)
void pointer_attention_fused(const float* __restrict__ src_embed,
                             const float* __restrict__ tgt_embed,
                             const float* __restrict__ tgt,
                             float* __restrict__ matched) {
  __shared__ __bf16 sQ[BLOCK_M][QSTR];
  __shared__ __bf16 sK[TILE_N][QSTR];
  __shared__ float  sV[TILE_N][4];

  const int b    = blockIdx.y;
  const int m0   = blockIdx.x * BLOCK_M;
  const int tid  = threadIdx.x;
  const int lane = tid & 31;
  const int wave = tid >> 5;
  const int half = lane >> 4;
  const int idx  = lane & 15;

  const float* qg = src_embed + ((size_t)b * Nn + m0) * Dn;
  #pragma unroll 4
  for (int i = 0; i < (BLOCK_M * Dn / 4) / THREADS; ++i) {
    int linear = i * THREADS + tid;
    int row  = linear >> 7;
    int quad = linear & 127;
    float4 x = *(const float4*)(qg + (size_t)row * Dn + quad * 4);
    uint2 u;
    u.x = pack2(x.x, x.y);
    u.y = pack2(x.z, x.w);
    *(uint2*)&sQ[row][quad * 4] = u;
  }

  float Mx[8], Ls[8], Ax[8], Ay[8], Az[8];
  #pragma unroll
  for (int j = 0; j < 8; ++j) { Mx[j] = -3.0e38f; Ls[j] = 0.f; Ax[j] = Ay[j] = Az[j] = 0.f; }

  const __bf16* qrow  = &sQ[wave * 16 + idx][0];
  const __bf16* krow0 = &sK[idx][0];
  const __bf16* krow1 = &sK[16 + idx][0];

  for (int kt = 0; kt < NTILES; ++kt) {
    const int n0 = kt * TILE_N;
    __syncthreads();

    const float* kg = tgt_embed + ((size_t)b * Nn + n0) * Dn;
    #pragma unroll 4
    for (int i = 0; i < (TILE_N * Dn / 4) / THREADS; ++i) {
      int linear = i * THREADS + tid;
      int row  = linear >> 7;
      int quad = linear & 127;
      float4 x = *(const float4*)(kg + (size_t)row * Dn + quad * 4);
      uint2 u;
      u.x = pack2(x.x, x.y);
      u.y = pack2(x.z, x.w);
      *(uint2*)&sK[row][quad * 4] = u;
    }
    if (tid < TILE_N) {
      const float* vp = tgt + ((size_t)b * Nn + n0 + tid) * 3;
      sV[tid][0] = vp[0]; sV[tid][1] = vp[1]; sV[tid][2] = vp[2];
    }
    {
      int nn = n0 + TILE_N; if (nn >= Nn) nn = 0;
      const char* nxt = (const char*)(tgt_embed + ((size_t)b * Nn + nn) * Dn);
      __builtin_prefetch(nxt + (size_t)tid * 512, 0, 1);
    }
    __syncthreads();

    v8f c0 = {};
    v8f c1 = {};
    #pragma unroll
    for (int kc = 0; kc < Dn / 32; ++kc) {
      const int k0 = kc * 32 + half * 8;
      bffrag a, b0, b1;
      a.h[0]  = *(const v8bf*)(qrow  + k0);
      a.h[1]  = *(const v8bf*)(qrow  + k0 + 16);
      b0.h[0] = *(const v8bf*)(krow0 + k0);
      b0.h[1] = *(const v8bf*)(krow0 + k0 + 16);
      b1.h[0] = *(const v8bf*)(krow1 + k0);
      b1.h[1] = *(const v8bf*)(krow1 + k0 + 16);
      c0 = __builtin_amdgcn_wmma_f32_16x16x32_bf16(false, a.v, false, b0.v, (short)0, c0, false, false);
      c1 = __builtin_amdgcn_wmma_f32_16x16x32_bf16(false, a.v, false, b1.v, (short)0, c1, false, false);
    }

    const float v00 = sV[idx][0],      v01 = sV[idx][1],      v02 = sV[idx][2];
    const float v10 = sV[16 + idx][0], v11 = sV[16 + idx][1], v12 = sV[16 + idx][2];
    #pragma unroll
    for (int j = 0; j < 8; ++j) {
      float s0 = c0[j] * SCALE_F;
      float s1 = c1[j] * SCALE_F;
      float mx   = redmax16(fmaxf(s0, s1));
      float mnew = fmaxf(Mx[j], mx);
      float fscl = __expf(Mx[j] - mnew);
      float p0 = __expf(s0 - mnew);
      float p1 = __expf(s1 - mnew);
      Ls[j] = Ls[j] * fscl + p0 + p1;
      Ax[j] = Ax[j] * fscl + p0 * v00 + p1 * v10;
      Ay[j] = Ay[j] * fscl + p0 * v01 + p1 * v11;
      Az[j] = Az[j] * fscl + p0 * v02 + p1 * v12;
      Mx[j] = mnew;
    }
  }

  #pragma unroll
  for (int j = 0; j < 8; ++j) {
    Ls[j] = redsum16(Ls[j]);
    Ax[j] = redsum16(Ax[j]);
    Ay[j] = redsum16(Ay[j]);
    Az[j] = redsum16(Az[j]);
  }
  if (idx == 0) {
    #pragma unroll
    for (int j = 0; j < 8; ++j) {
      int row = m0 + wave * 16 + j + 8 * half;
      float inv = 1.0f / Ls[j];
      float* mp = matched + ((size_t)b * Nn + row) * 3;
      mp[0] = Ax[j] * inv;
      mp[1] = Ay[j] * inv;
      mp[2] = Az[j] * inv;
    }
  }
}

// ---------------------------------------------------------------------------
// Per-batch: moments -> H -> 3x3 SVD (Jacobi on H^T H) -> R, t
// ---------------------------------------------------------------------------
__global__ __launch_bounds__(256)
void procrustes_kernel(const float* __restrict__ src,
                       const float* __restrict__ matched,
                       float* __restrict__ out) {
  __shared__ float red[256];
  const int b = blockIdx.x, tid = threadIdx.x;

  float acc[15];
  #pragma unroll
  for (int k = 0; k < 15; ++k) acc[k] = 0.f;
  for (int n = tid; n < Nn; n += 256) {
    const float* sp = src     + ((size_t)b * Nn + n) * 3;
    const float* mp = matched + ((size_t)b * Nn + n) * 3;
    float sx = sp[0], sy = sp[1], sz = sp[2];
    float mx = mp[0], my = mp[1], mz = mp[2];
    acc[0] += sx; acc[1] += sy; acc[2] += sz;
    acc[3] += mx; acc[4] += my; acc[5] += mz;
    acc[6]  += sx * mx; acc[7]  += sx * my; acc[8]  += sx * mz;
    acc[9]  += sy * mx; acc[10] += sy * my; acc[11] += sy * mz;
    acc[12] += sz * mx; acc[13] += sz * my; acc[14] += sz * mz;
  }

  float tot[15];
  for (int k = 0; k < 15; ++k) {
    red[tid] = acc[k];
    __syncthreads();
    for (int off = 128; off > 0; off >>= 1) {
      if (tid < off) red[tid] += red[tid + off];
      __syncthreads();
    }
    tot[k] = red[0];
    __syncthreads();
  }

  if (tid == 0) {
    const float invN = 1.0f / (float)Nn;
    float sc[3] = { tot[0] * invN, tot[1] * invN, tot[2] * invN };
    float mc[3] = { tot[3] * invN, tot[4] * invN, tot[5] * invN };
    float H[3][3];
    for (int r = 0; r < 3; ++r)
      for (int c = 0; c < 3; ++c)
        H[r][c] = tot[6 + r * 3 + c] - (float)Nn * sc[r] * mc[c];

    float A[3][3];
    for (int i = 0; i < 3; ++i)
      for (int j = 0; j < 3; ++j)
        A[i][j] = H[0][i] * H[0][j] + H[1][i] * H[1][j] + H[2][i] * H[2][j];
    float V[3][3] = { {1.f,0.f,0.f}, {0.f,1.f,0.f}, {0.f,0.f,1.f} };
    const int PQ[3][2] = { {0,1}, {0,2}, {1,2} };
    for (int sweep = 0; sweep < 12; ++sweep) {
      for (int e = 0; e < 3; ++e) {
        int p = PQ[e][0], q = PQ[e][1];
        float apq = A[p][q];
        if (fabsf(apq) < 1e-30f) continue;
        float theta = (A[q][q] - A[p][p]) / (2.f * apq);
        float t = (theta >= 0.f ? 1.f : -1.f) / (fabsf(theta) + sqrtf(theta * theta + 1.f));
        float cc = 1.f / sqrtf(t * t + 1.f);
        float ss = t * cc;
        for (int k = 0; k < 3; ++k) { float akp = A[k][p], akq = A[k][q];
          A[k][p] = cc * akp - ss * akq; A[k][q] = ss * akp + cc * akq; }
        for (int k = 0; k < 3; ++k) { float apk = A[p][k], aqk = A[q][k];
          A[p][k] = cc * apk - ss * aqk; A[q][k] = ss * apk + cc * aqk; }
        for (int k = 0; k < 3; ++k) { float vkp = V[k][p], vkq = V[k][q];
          V[k][p] = cc * vkp - ss * vkq; V[k][q] = ss * vkp + cc * vkq; }
      }
    }

    float U[3][3];
    for (int i = 0; i < 3; ++i) {
      float u0 = H[0][0] * V[0][i] + H[0][1] * V[1][i] + H[0][2] * V[2][i];
      float u1 = H[1][0] * V[0][i] + H[1][1] * V[1][i] + H[1][2] * V[2][i];
      float u2 = H[2][0] * V[0][i] + H[2][1] * V[1][i] + H[2][2] * V[2][i];
      float nrm = sqrtf(u0 * u0 + u1 * u1 + u2 * u2);
      float inv = 1.f / fmaxf(nrm, 1e-20f);
      U[0][i] = u0 * inv; U[1][i] = u1 * inv; U[2][i] = u2 * inv;
    }

    float R[3][3];
    for (int r = 0; r < 3; ++r)
      for (int c = 0; c < 3; ++c)
        R[r][c] = V[r][0] * U[c][0] + V[r][1] * U[c][1] + V[r][2] * U[c][2];
    float det = R[0][0] * (R[1][1] * R[2][2] - R[1][2] * R[2][1])
              - R[0][1] * (R[1][0] * R[2][2] - R[1][2] * R[2][0])
              + R[0][2] * (R[1][0] * R[2][1] - R[1][1] * R[2][0]);
    if (det < 0.f) { R[2][0] = -R[2][0]; R[2][1] = -R[2][1]; R[2][2] = -R[2][2]; }

    float tv[3];
    for (int c = 0; c < 3; ++c)
      tv[c] = -(R[c][0] * sc[0] + R[c][1] * sc[1] + R[c][2] * sc[2]) + mc[c];

    for (int r = 0; r < 3; ++r)
      for (int c = 0; c < 3; ++c)
        out[(size_t)b * 9 + r * 3 + c] = R[r][c];
    out[(size_t)Bn * 9 + b * 3 + 0] = tv[0];
    out[(size_t)Bn * 9 + b * 3 + 1] = tv[1];
    out[(size_t)Bn * 9 + b * 3 + 2] = tv[2];
  }
}

extern "C" void kernel_launch(void* const* d_in, const int* in_sizes, int n_in,
                              void* d_out, int out_size, void* d_ws, size_t ws_size,
                              hipStream_t stream) {
  const float* src       = (const float*)d_in[0];   // (B, N, 3)
  const float* tgt       = (const float*)d_in[1];   // (B, N, 3)
  const float* src_embed = (const float*)d_in[2];   // (B, N, D)
  const float* tgt_embed = (const float*)d_in[3];   // (B, N, D)
  float* out = (float*)d_out;                       // R (B,3,3) then t (B,1,3)

  const size_t WS_MATCHED = 0;                          // (B,N,3) f32 = 768 KB
  const size_t WS_QBF     = 1u << 20;                   // bf16 mirror of src_embed (64 MB)
  const size_t EMB_BYTES  = (size_t)Bn * Nn * Dn * 2;
  const size_t WS_KBF     = WS_QBF + EMB_BYTES;         // bf16 mirror of tgt_embed (64 MB)
  const size_t WS_NEED    = WS_KBF + EMB_BYTES;

  float* matched = (float*)((char*)d_ws + WS_MATCHED);
  dim3 gridA(Nn / BLOCK_M, Bn);

  if (ws_size >= WS_NEED) {
    __bf16* qbf = (__bf16*)((char*)d_ws + WS_QBF);
    __bf16* kbf = (__bf16*)((char*)d_ws + WS_KBF);
    const int nquads = Bn * Nn * Dn / 4;               // 8,388,608
    convert_f32_bf16<<<(nquads + 255) / 256, 256, 0, stream>>>(src_embed, (uint2*)qbf, nquads);
    convert_f32_bf16<<<(nquads + 255) / 256, 256, 0, stream>>>(tgt_embed, (uint2*)kbf, nquads);
    pointer_attention_async<<<gridA, THREADS, 0, stream>>>(qbf, kbf, tgt, matched);
  } else {
    pointer_attention_fused<<<gridA, THREADS, 0, stream>>>(src_embed, tgt_embed, tgt, matched);
  }
  procrustes_kernel<<<Bn, 256, 0, stream>>>(src, matched, out);
}
